// LoraAdapterLayer_49520972922883
// MI455X (gfx1250) — matrix operand order repaired and checked by reading, not compiled, not run on hardware
//
#include <hip/hip_runtime.h>

// LoRA layer fused into a single f16-WMMA GEMM:
//   Weff[n][k] = W[n][k] + 2 * sum_r A[k][r]*B[r][n]   (stored f16, [N][K])
//   out[m][n]  = sum_k x_f16[m][k]*Weff[n][k] + b[n]   (f32 accumulate)

#define TOKENS 8192
#define DIM    4096
#define DIMOUT 4096
#define RANK   16
#define LSCALE 2.0f

typedef __attribute__((ext_vector_type(16))) _Float16 v16h;
typedef __attribute__((ext_vector_type(8)))  _Float16 v8h;
typedef __attribute__((ext_vector_type(8)))  float    v8f;
typedef __attribute__((ext_vector_type(4)))  float    v4f;

// ---- gfx1250 async-copy path (guarded; falls back to ld/st staging) ----
#if __has_builtin(__builtin_amdgcn_global_load_async_to_lds_b128)
#define USE_ASYNC 1
typedef int v4i_vs __attribute__((vector_size(16)));
typedef __attribute__((address_space(1))) v4i_vs* g_v4i_p;
typedef __attribute__((address_space(3))) v4i_vs* l_v4i_p;
__device__ __forceinline__ void async_cp16(const void* g, void* l) {
  __builtin_amdgcn_global_load_async_to_lds_b128((g_v4i_p)g, (l_v4i_p)l, 0, 0);
}
#if __has_builtin(__builtin_amdgcn_s_wait_asynccnt)
#define WAIT_ASYNC0() __builtin_amdgcn_s_wait_asynccnt(0)
#else
#define WAIT_ASYNC0() asm volatile("s_wait_asynccnt 0x0" ::: "memory")
#endif
#else
#define USE_ASYNC 0
#endif

// ------------------------------------------------------------------
// Kernel 1: convert x (f32) -> f16, 8 elements/thread, b128 traffic.
// x is read once -> non-temporal loads; xh is re-read by GEMM -> RT store.
// ------------------------------------------------------------------
__global__ __launch_bounds__(256) void k_cvt_x(const float* __restrict__ x,
                                               _Float16* __restrict__ xh) {
  size_t i = ((size_t)blockIdx.x * 256 + threadIdx.x) * 8;
  v4f a = __builtin_nontemporal_load((const v4f*)(x + i));
  v4f b = __builtin_nontemporal_load((const v4f*)(x + i) + 1);
  v8h h;
  h[0] = (_Float16)a.x; h[1] = (_Float16)a.y;
  h[2] = (_Float16)a.z; h[3] = (_Float16)a.w;
  h[4] = (_Float16)b.x; h[5] = (_Float16)b.y;
  h[6] = (_Float16)b.z; h[7] = (_Float16)b.w;
  *(v8h*)(xh + i) = h;
}

// ------------------------------------------------------------------
// Kernel 2: Weff^T = W + 2*(A@B)^T, stored f16 row-major [N][K].
// Block = (n, K-half). B-column (16 f32) staged in LDS once.
// ------------------------------------------------------------------
__global__ __launch_bounds__(256) void k_weff(const float* __restrict__ W,
                                              const float* __restrict__ A,
                                              const float* __restrict__ B,
                                              _Float16* __restrict__ wt) {
  const int n  = blockIdx.x;   // output column
  const int kh = blockIdx.y;   // which half of K
  __shared__ float bcol[RANK];
  if (threadIdx.x < RANK)
    bcol[threadIdx.x] = B[(size_t)threadIdx.x * DIMOUT + n];
  __syncthreads();
  float bc[RANK];
#pragma unroll
  for (int r = 0; r < RANK; ++r) bc[r] = bcol[r];
#pragma unroll
  for (int i = 0; i < 8; ++i) {
    int k = kh * 2048 + i * 256 + threadIdx.x;
    const v4f* ar = (const v4f*)(A + (size_t)k * RANK);
    float s = 0.0f;
#pragma unroll
    for (int q = 0; q < 4; ++q) {
      v4f av = __builtin_nontemporal_load(ar + q);
      s += av.x * bc[q * 4 + 0] + av.y * bc[q * 4 + 1] +
           av.z * bc[q * 4 + 2] + av.w * bc[q * 4 + 3];
    }
    float w = __builtin_nontemporal_load(W + (size_t)n * DIM + k) + LSCALE * s;
    wt[(size_t)n * DIM + k] = (_Float16)w;
  }
}

// ------------------------------------------------------------------
// Kernel 3: GEMM out = x_f16 * Weff^T + b  using v_wmma_f32_16x16x32_f16
// Block tile 128(M) x 128(N), K-step 32, 8 waves (4 along M x 2 along N),
// wave tile 32x64 -> 2x4 accumulators of 16x16 f32 (8 VGPRs each).
// Double-buffered LDS, tiles staged via global_load_async_to_lds_b128.
// ------------------------------------------------------------------
#define BM 128
#define BN 128
#define BK 32
#define NT (DIM / BK)   // 128 K-tiles
#define LDS_STRIDE 80   // 32 halfs = 64B + 16B pad (16B aligned, rotates banks)
#define TILEPAIR (2 * BM * LDS_STRIDE)  // X tile + W tile = 20 KB

__global__ __launch_bounds__(256) void k_gemm(const _Float16* __restrict__ xh,
                                              const _Float16* __restrict__ wt,
                                              const float* __restrict__ bias,
                                              float* __restrict__ out) {
  __shared__ alignas(16) unsigned char lds[2 * TILEPAIR]; // 40 KB double-buffered

  const int tid  = threadIdx.x;
  const int lane = tid & 31;
  const int wave = tid >> 5;
  const int l15  = lane & 15;
  const int half = lane >> 4;

  const int m0 = blockIdx.y * BM;
  const int n0 = blockIdx.x * BN;
  const int waveM = wave & 3;   // 4 waves along M, 32 rows each
  const int waveN = wave >> 2;  // 2 waves along N, 64 cols each

  // Per-thread staging chunks: 128 rows x 64B = 512 x 16B chunks, 2/thread
  const _Float16* gx[2];
  const _Float16* gw[2];
  unsigned loff[2];
#pragma unroll
  for (int t = 0; t < 2; ++t) {
    int c   = t * 256 + tid;
    int row = c >> 2;
    int col = c & 3;
    gx[t]   = xh + (size_t)(m0 + row) * DIM + col * 8;
    gw[t]   = wt + (size_t)(n0 + row) * DIM + col * 8;
    loff[t] = row * LDS_STRIDE + col * 16;
  }

  const v8f vzero = {0.f, 0.f, 0.f, 0.f, 0.f, 0.f, 0.f, 0.f};
  v8f acc[2][4];
#pragma unroll
  for (int i = 0; i < 2; ++i)
#pragma unroll
    for (int j = 0; j < 4; ++j) acc[i][j] = vzero;

#if USE_ASYNC
  auto issue_tile = [&](int kt, int buf) {
    unsigned char* bX = lds + buf * TILEPAIR;
    unsigned char* bW = bX + BM * LDS_STRIDE;
#pragma unroll
    for (int t = 0; t < 2; ++t) {
      async_cp16(gx[t] + (size_t)kt * BK, bX + loff[t]);
      async_cp16(gw[t] + (size_t)kt * BK, bW + loff[t]);
    }
  };
  issue_tile(0, 0);
#endif

  for (int kt = 0; kt < NT; ++kt) {
    const int cur = kt & 1;
    unsigned char* ldsX;
    unsigned char* ldsW;
#if USE_ASYNC
    ldsX = lds + cur * TILEPAIR;
    ldsW = ldsX + BM * LDS_STRIDE;
    WAIT_ASYNC0();        // this wave's tile-kt copies have landed in LDS
    __syncthreads();      // all waves' copies landed; all done reading buf
    if (kt + 1 < NT) issue_tile(kt + 1, 1 - cur);  // overlap with compute
#else
    ldsX = lds;
    ldsW = lds + BM * LDS_STRIDE;
    __syncthreads();      // protect LDS reads of previous iteration
#pragma unroll
    for (int t = 0; t < 2; ++t) {
      *(uint4*)(ldsX + loff[t]) = *(const uint4*)(gx[t] + (size_t)kt * BK);
      *(uint4*)(ldsW + loff[t]) = *(const uint4*)(gw[t] + (size_t)kt * BK);
      if (kt + 1 < NT) {
        __builtin_prefetch((const char*)(gx[t] + (size_t)(kt + 1) * BK), 0, 3);
        __builtin_prefetch((const char*)(gw[t] + (size_t)(kt + 1) * BK), 0, 3);
      }
    }
    __syncthreads();
#endif

    // A fragments (16x32 f16): lane<16 holds K0-7 & K16-23, lane>=16 K8-15 & K24-31
    v16h afrag[2], bfrag[4];
#pragma unroll
    for (int i = 0; i < 2; ++i) {
      int row = waveM * 32 + i * 16 + l15;
      const unsigned char* p = ldsX + row * LDS_STRIDE + half * 16;
      union { uint4 q[2]; v16h v; } u;
      u.q[0] = *(const uint4*)p;
      u.q[1] = *(const uint4*)(p + 32);
      afrag[i] = u.v;
    }
    // B fragments (32x16 f16): lanes 0-15 col n,K0-7 & K16-23; lanes 16-31 col n,K8-15 & K24-31
#pragma unroll
    for (int j = 0; j < 4; ++j) {
      int row = waveN * 64 + j * 16 + l15;   // row of Weff^T == output column
      const unsigned char* p = ldsW + row * LDS_STRIDE + half * 16;
      union { uint4 q[2]; v16h v; } u;
      u.q[0] = *(const uint4*)p;
      u.q[1] = *(const uint4*)(p + 32);
      bfrag[j] = u.v;
    }
#pragma unroll
    for (int i = 0; i < 2; ++i)
#pragma unroll
      for (int j = 0; j < 4; ++j)
        acc[i][j] = __builtin_amdgcn_wmma_f32_16x16x32_f16(
            false, afrag[i], false, bfrag[j], (short)0, acc[i][j], false, false);
  }

  // Epilogue: C/D layout — VGPR v: lanes0-15 -> M=v, lanes16-31 -> M=v+8.
  // Output is a 134MB stream -> non-temporal stores (don't evict x/W from L2).
#pragma unroll
  for (int j = 0; j < 4; ++j) {
    int nn = n0 + waveN * 64 + j * 16 + l15;
    float bv = bias[nn];
#pragma unroll
    for (int i = 0; i < 2; ++i) {
      int mbase = m0 + waveM * 32 + i * 16 + half * 8;
#pragma unroll
      for (int v = 0; v < 8; ++v)
        __builtin_nontemporal_store(acc[i][j][v] + bv,
                                    out + (size_t)(mbase + v) * DIMOUT + nn);
    }
  }
}

// ------------------------------------------------------------------
extern "C" void kernel_launch(void* const* d_in, const int* in_sizes, int n_in,
                              void* d_out, int out_size, void* d_ws, size_t ws_size,
                              hipStream_t stream) {
  const float* x = (const float*)d_in[0];
  const float* A = (const float*)d_in[1];
  const float* B = (const float*)d_in[2];
  const float* W = (const float*)d_in[3];
  const float* b = (const float*)d_in[4];
  float* out = (float*)d_out;

  // Workspace: x_f16 (64 MB) then Weff_f16 (32 MB)
  _Float16* xh = (_Float16*)d_ws;
  _Float16* wtp = (_Float16*)((char*)d_ws + (size_t)TOKENS * DIM * sizeof(_Float16));

  k_cvt_x<<<(TOKENS * DIM / 8 + 255) / 256, 256, 0, stream>>>(x, xh);
  k_weff<<<dim3(DIMOUT, DIM / 2048), 256, 0, stream>>>(W, A, B, wtp);
  k_gemm<<<dim3(DIMOUT / BN, TOKENS / BM), 256, 0, stream>>>(xh, wtp, b, out);
}
